// SliceAttention_85375359910283
// MI455X (gfx1250) — matrix operand order repaired and checked by reading, not compiled, not run on hardware
//
#include <hip/hip_runtime.h>
#include <hip/hip_bf16.h>
#include <math.h>

typedef __attribute__((ext_vector_type(16))) __bf16 v16bf;
typedef __attribute__((ext_vector_type(8)))  float  v8f;

#define DEV __device__ __forceinline__

constexpr int  Bc = 8, Nc = 16384, HIDc = 256, NHc = 8, NSc = 32;

// ---- d_out layout (floats, reference return order) ----
constexpr long OUT_OFF  = 0;                             // (B,N,HID)
constexpr long SW_OFF   = (long)Bc * Nc * HIDc;          // slice_weights (B,NH,NS,N)
constexpr long TEMP_OFF = SW_OFF + (long)Bc*NHc*NSc*Nc;  // temp (8)
constexpr long BIAS_OFF = TEMP_OFF + NHc;                // bias (256)
constexpr long ATTN_OFF = BIAS_OFF + (long)NHc*NSc;      // attn (B,NH,32,32)

// ---- workspace layout (byte offsets) ----
constexpr size_t KVBF_B  = 0;                      // kv bf16 (B,N,512)            134 MB
constexpr size_t SCT_B   = 134217728;              // scoresT f32 (B,NH,N,32)      134 MB
constexpr size_t OPBF_B  = 134217728;              // out_pre bf16 alias (dead scoresT)
constexpr size_t WBF_B   = 268435456;              // w bf16 (B,NH,NS,N)            67 MB
constexpr size_t NORM_B  = 335544320;              // slice_norm f32 (64*32)
constexpr size_t OT_B    = 335552512;              // out_token f32 (64*32*32)
constexpr size_t WKVWB_B = 335814656;              // wkv_w bf16 (512*256)
constexpr size_t OUTWB_B = 336076800;              // out_w bf16 (256*256)

// ================= WMMA tile helpers (wave32, 16x16x32 bf16) =================
// A (16x32, M x K): lane l -> row=l&15 ; elem e -> K = ((l>>4)<<4) + e
DEV v16bf load_a_f32(const float* src, long srow, long sk) {
  int lane = threadIdx.x & 31;
  int row  = lane & 15;
  int kb   = (lane >> 4) << 4;
  v16bf a;
#pragma unroll
  for (int e = 0; e < 16; ++e) {
    int k = kb + e;
    a[e] = (__bf16)src[(long)row * srow + (long)k * sk];
  }
  return a;
}
DEV v16bf load_a_bf(const __bf16* src, long srow, long sk) {
  int lane = threadIdx.x & 31;
  int row  = lane & 15;
  int kb   = (lane >> 4) << 4;
  v16bf a;
#pragma unroll
  for (int e = 0; e < 16; ++e) {
    int k = kb + e;
    a[e] = src[(long)row * srow + (long)k * sk];
  }
  return a;
}
// B (32x16, K x N): lane l -> col=l&15 ; elem e -> K = ((l>>4)<<4) + e
DEV v16bf load_b_f32(const float* src, long sk, long sn) {
  int lane = threadIdx.x & 31;
  int col  = lane & 15;
  int kb   = (lane >> 4) << 4;
  v16bf b;
#pragma unroll
  for (int e = 0; e < 16; ++e)
    b[e] = (__bf16)src[(long)(kb + e) * sk + (long)col * sn];
  return b;
}
DEV v16bf load_b_bf(const __bf16* src, long sk, long sn) {
  int lane = threadIdx.x & 31;
  int col  = lane & 15;
  int kb   = (lane >> 4) << 4;
  v16bf b;
#pragma unroll
  for (int e = 0; e < 16; ++e)
    b[e] = src[(long)(kb + e) * sk + (long)col * sn];
  return b;
}
// C/D (16x16 f32): vgpr r, lanes0-15 -> M=r ; lanes16-31 -> M=r+8 ; N=lane&15
DEV void store_tile(float* dst, long sm, long sn, const v8f& acc, float bias) {
  int lane = threadIdx.x & 31;
  int col  = lane & 15;
  int rb   = (lane >> 4) << 3;
#pragma unroll
  for (int r = 0; r < 8; ++r)
    dst[(long)(rb + r) * sm + (long)col * sn] = acc[r] + bias;
}
DEV void store_tile_bf(__bf16* dst, long sm, long sn, const v8f& acc, float bias) {
  int lane = threadIdx.x & 31;
  int col  = lane & 15;
  int rb   = (lane >> 4) << 3;
#pragma unroll
  for (int r = 0; r < 8; ++r)
    dst[(long)(rb + r) * sm + (long)col * sn] = (__bf16)(acc[r] + bias);
}
DEV v8f zero8() {
  v8f z;
#pragma unroll
  for (int e = 0; e < 8; ++e) z[e] = 0.f;
  return z;
}

// ======== K0: pre-convert weight matrices to bf16 (one pass, tiny) ========
__global__ void k0_cvt_w(const float* __restrict__ wkv_w, const float* __restrict__ out_w,
                         __bf16* __restrict__ wkvw_bf, __bf16* __restrict__ outw_bf) {
  int i = blockIdx.x * 256 + threadIdx.x;
  if (i < 512 * 256) wkvw_bf[i] = (__bf16)wkv_w[i];
  if (i < 256 * 256) outw_bf[i] = (__bf16)out_w[i];
}

// ============ K1: kv = x @ wkv_w^T + wkv_b   (131072x512, K=256) ============
// block = 8 waves sharing one 16-row A strip; each wave owns 4 n-tiles.
__global__ void k1_kv(const float* __restrict__ x, const __bf16* __restrict__ wkvw_bf,
                      const float* __restrict__ wkv_b, __bf16* __restrict__ kv_bf) {
  int wave = threadIdx.x >> 5;
  long task = (long)blockIdx.x * 8 + wave;       // tm*8 + q
  long tm = task >> 3;                           // 0..8191
  int  q  = (int)(task & 7);
  int  tn0 = q * 4;                              // 4 of 32 n-tiles
  v8f acc[4];
#pragma unroll
  for (int j = 0; j < 4; ++j) acc[j] = zero8();
  const float* xrow = x + tm * 16 * 256;
#pragma unroll 1
  for (int kc = 0; kc < 8; ++kc) {
    __builtin_prefetch(xrow + (kc + 1) * 32, 0, 1);
    v16bf a = load_a_f32(xrow + kc * 32, 256, 1);
#pragma unroll
    for (int j = 0; j < 4; ++j) {
      v16bf b = load_b_bf(wkvw_bf + (long)(tn0 + j) * 16 * 256 + kc * 32, 1, 256);
      acc[j] = __builtin_amdgcn_wmma_f32_16x16x32_bf16(false, a, false, b,
                                                       (short)0, acc[j], false, false);
    }
  }
  int lane = threadIdx.x & 31, col = lane & 15;
#pragma unroll
  for (int j = 0; j < 4; ++j) {
    float bias = wkv_b[(tn0 + j) * 16 + col];
    store_tile_bf(kv_bf + tm * 16 * 512 + (long)(tn0 + j) * 16, 512, 1, acc[j], bias);
  }
}

// ==== K2: scoresT[b,h,n,m] = wtq[h]·xk + bias  (per bh: M=32,N=16384,K=32) ====
__global__ void k2_scores(const __bf16* __restrict__ kv_bf, const float* __restrict__ wtq,
                          const float* __restrict__ wtq_bias, float* __restrict__ scT) {
  int wave = threadIdx.x >> 5;
  long task = (long)blockIdx.x * 8 + wave;       // bh*1024 + ntile
  int bh = (int)(task >> 10);
  int nt = (int)(task & 1023);
  int b = bh >> 3, h = bh & 7;
  long n0 = (long)nt * 16;
  int lane = threadIdx.x & 31, rb = (lane >> 4) << 3;
  // B tile: B[k=d, n=tok] = kv[(b*N+n0+tok)*512 + h*32 + d]
  v16bf bt = load_b_bf(kv_bf + ((long)b * Nc + n0) * 512 + h * 32, 1, 512);
#pragma unroll
  for (int mt = 0; mt < 2; ++mt) {
    int m0 = mt * 16;
    v16bf a = load_a_f32(wtq + (long)h * 32 * 32 + m0 * 32, 32, 1);
    v8f c;
#pragma unroll
    for (int r = 0; r < 8; ++r) c[r] = wtq_bias[h * 32 + m0 + rb + r];
    c = __builtin_amdgcn_wmma_f32_16x16x32_bf16(false, a, false, bt,
                                                (short)0, c, false, false);
    // transposed store: scoresT[(bh*N + n)*32 + m]
    store_tile(scT + ((long)bh * Nc + n0) * 32 + m0, 1, 32, c, 0.f);
  }
}

// ======== K3: per-token top-8-of-32 + softmax -> slice_weights, norms ========
__global__ void k3_topk(const float* __restrict__ scT, float* __restrict__ swq,
                        __bf16* __restrict__ w_bf, float* __restrict__ g_norm) {
  __shared__ float nrm[NSc];
  int tid = threadIdx.x;
  long gid = (long)blockIdx.x * 256 + tid;       // bh*N + n
  int bh = (int)(gid >> 14);
  int n  = (int)(gid & (Nc - 1));
  if (tid < NSc) nrm[tid] = 0.f;
  __syncthreads();

  float s[NSc];
  const float* src = scT + gid * NSc;
#pragma unroll
  for (int m = 0; m < NSc; ++m) s[m] = src[m];
  // exact top-8 selection (first-index tie break, matches lax.top_k)
  unsigned sel = 0u;
#pragma unroll 1
  for (int kk = 0; kk < 8; ++kk) {
    float best = -INFINITY; int bi = 0;
#pragma unroll
    for (int m = 0; m < NSc; ++m)
      if (!((sel >> m) & 1u) && s[m] > best) { best = s[m]; bi = m; }
    sel |= 1u << bi;
  }
  float mx = -INFINITY;
#pragma unroll
  for (int m = 0; m < NSc; ++m) if ((sel >> m) & 1u) mx = fmaxf(mx, s[m]);
  float sum = 0.f; float w[NSc];
#pragma unroll
  for (int m = 0; m < NSc; ++m) {
    float e = ((sel >> m) & 1u) ? __expf(s[m] - mx) : 0.f;  // exp(-1e6-mx) flushes to 0
    w[m] = e; sum += e;
  }
  float inv = 1.f / sum;
  float*  dstf = swq  + (long)bh * NSc * Nc + n;
  __bf16* dstb = w_bf + (long)bh * NSc * Nc + n;
#pragma unroll
  for (int m = 0; m < NSc; ++m) {
    float wm = w[m] * inv;
    dstf[(long)m * Nc] = wm;                     // fp32 output (coalesced over n)
    dstb[(long)m * Nc] = (__bf16)wm;             // bf16 shadow for K4/K5
    atomicAdd(&nrm[m], wm);
  }
  __syncthreads();
  if (tid < NSc) atomicAdd(&g_norm[bh * NSc + tid], nrm[tid]);
}

// ==== K4: slice_token = w·xv (WMMA, K split over 8 waves), then tiny attn ====
__global__ void k4_slice(const __bf16* __restrict__ kv_bf, const __bf16* __restrict__ w_bf,
                         const float* __restrict__ g_norm, const float* __restrict__ qp,
                         float* __restrict__ ot, float* __restrict__ attn_out) {
  __shared__ float st[32 * 32];
  __shared__ float qkvs[32 * 96];
  __shared__ float dotsS[32 * 32];
  __shared__ float attnS[32 * 32];
  int bh = blockIdx.x, b = bh >> 3, h = bh & 7;
  int tid = threadIdx.x, wave = tid >> 5, lane = tid & 31;
  for (int i = tid; i < 1024; i += 256) st[i] = 0.f;
  __syncthreads();

  // phase 1: acc[mt][dt] += w[m-tile, 32 toks] x xv[32 toks, d-tile]
  v8f acc[2][2];
#pragma unroll
  for (int i = 0; i < 2; ++i)
#pragma unroll
    for (int j = 0; j < 2; ++j) acc[i][j] = zero8();
#pragma unroll 1
  for (int c = wave; c < Nc / 32; c += 8) {
    long nb = (long)c * 32;
    v16bf a[2], bm[2];
#pragma unroll
    for (int mt = 0; mt < 2; ++mt)
      a[mt] = load_a_bf(w_bf + ((long)bh * 32 + mt * 16) * Nc + nb, Nc, 1);
#pragma unroll
    for (int dt = 0; dt < 2; ++dt)
      bm[dt] = load_b_bf(kv_bf + ((long)b * Nc + nb) * 512 + 256 + h * 32 + dt * 16, 512, 1);
#pragma unroll
    for (int mt = 0; mt < 2; ++mt)
#pragma unroll
      for (int dt = 0; dt < 2; ++dt)
        acc[mt][dt] = __builtin_amdgcn_wmma_f32_16x16x32_bf16(false, a[mt], false, bm[dt],
                                                              (short)0, acc[mt][dt], false, false);
  }
  {
    int col = lane & 15, rb = (lane >> 4) << 3;
#pragma unroll
    for (int mt = 0; mt < 2; ++mt)
#pragma unroll
      for (int dt = 0; dt < 2; ++dt)
#pragma unroll
        for (int r = 0; r < 8; ++r)
          atomicAdd(&st[(mt * 16 + rb + r) * 32 + dt * 16 + col], acc[mt][dt][r]);
  }
  __syncthreads();
  // phase 2: normalize by slice_norm
  for (int i = tid; i < 1024; i += 256) {
    int m = i >> 5;
    st[i] = st[i] / (g_norm[bh * 32 + m] + 1e-5f);
  }
  __syncthreads();
  // phase 3: qkv = st @ qkv_proj[h]  (32x32 @ 32x96)
  for (int i = tid; i < 3072; i += 256) {
    int m = i / 96, col = i % 96;
    float acc2 = 0.f;
#pragma unroll
    for (int d = 0; d < 32; ++d)
      acc2 += st[m * 32 + d] * qp[((long)h * 32 + d) * 96 + col];
    qkvs[i] = acc2;
  }
  __syncthreads();
  // phase 4: dots = q k^T * scale
  const float scale = 0.17677669529663687f;     // 32^-0.5
  for (int i = tid; i < 1024; i += 256) {
    int r = i >> 5, cc = i & 31;
    float acc2 = 0.f;
#pragma unroll
    for (int d = 0; d < 32; ++d)
      acc2 += qkvs[r * 96 + d] * qkvs[cc * 96 + 32 + d];
    dotsS[i] = acc2 * scale;
  }
  __syncthreads();
  if (tid < 32) {
    float mx = -INFINITY;
#pragma unroll
    for (int j = 0; j < 32; ++j) mx = fmaxf(mx, dotsS[tid * 32 + j]);
    float sum = 0.f;
#pragma unroll
    for (int j = 0; j < 32; ++j) { float e = __expf(dotsS[tid * 32 + j] - mx); attnS[tid * 32 + j] = e; sum += e; }
    float inv = 1.f / sum;
#pragma unroll
    for (int j = 0; j < 32; ++j) {
      float av = attnS[tid * 32 + j] * inv;
      attnS[tid * 32 + j] = av;
      attn_out[((long)bh * 32 + tid) * 32 + j] = av;
    }
  }
  __syncthreads();
  // out_token = attn @ v
  for (int i = tid; i < 1024; i += 256) {
    int r = i >> 5, d = i & 31;
    float acc2 = 0.f;
#pragma unroll
    for (int j = 0; j < 32; ++j)
      acc2 += attnS[r * 32 + j] * qkvs[j * 96 + 64 + d];
    ot[((long)bh * 32 + r) * 32 + d] = acc2;
  }
}

// == K5: out_pre[b,n,h*32+d] = sum_m w[bh,m,n]*ot[bh,m,d]  (M=N tok, K=32, N=32) ==
__global__ void k5_outpre(const __bf16* __restrict__ w_bf, const float* __restrict__ ot,
                          __bf16* __restrict__ op_bf) {
  int wave = threadIdx.x >> 5;
  long task = (long)blockIdx.x * 8 + wave;       // bh*1024 + ttile
  int bh = (int)(task >> 10);
  int nt = (int)(task & 1023);
  int b = bh >> 3, h = bh & 7;
  long n0 = (long)nt * 16;
  // A[tok, k=m] = w_bf[(bh*32+m)*N + n0+tok]
  v16bf a = load_a_bf(w_bf + (long)bh * 32 * Nc + n0, 1, Nc);
#pragma unroll
  for (int dt = 0; dt < 2; ++dt) {
    v16bf bm = load_b_f32(ot + (long)bh * 1024 + dt * 16, 32, 1);
    v8f c = zero8();
    c = __builtin_amdgcn_wmma_f32_16x16x32_bf16(false, a, false, bm,
                                                (short)0, c, false, false);
    store_tile_bf(op_bf + ((long)b * Nc + n0) * 256 + h * 32 + dt * 16, 256, 1, c, 0.f);
  }
}

// ======== K6: out = out_pre @ out_w^T + out_b  (131072x256, K=256) ========
__global__ void k6_out(const __bf16* __restrict__ op_bf, const __bf16* __restrict__ outw_bf,
                       const float* __restrict__ out_b, float* __restrict__ out) {
  int wave = threadIdx.x >> 5;
  long task = (long)blockIdx.x * 8 + wave;       // tm*4 + q
  long tm = task >> 2;
  int  q  = (int)(task & 3);
  int  tn0 = q * 4;                              // 4 of 16 n-tiles
  v8f acc[4];
#pragma unroll
  for (int j = 0; j < 4; ++j) acc[j] = zero8();
  const __bf16* arow = op_bf + tm * 16 * 256;
#pragma unroll 1
  for (int kc = 0; kc < 8; ++kc) {
    __builtin_prefetch(arow + (kc + 1) * 32, 0, 1);
    v16bf a = load_a_bf(arow + kc * 32, 256, 1);
#pragma unroll
    for (int j = 0; j < 4; ++j) {
      v16bf b = load_b_bf(outw_bf + (long)(tn0 + j) * 16 * 256 + kc * 32, 1, 256);
      acc[j] = __builtin_amdgcn_wmma_f32_16x16x32_bf16(false, a, false, b,
                                                       (short)0, acc[j], false, false);
    }
  }
  int lane = threadIdx.x & 31, col = lane & 15;
#pragma unroll
  for (int j = 0; j < 4; ++j) {
    float bias = out_b[(tn0 + j) * 16 + col];
    store_tile(out + tm * 16 * 256 + (long)(tn0 + j) * 16, 256, 1, acc[j], bias);
  }
}

// ================= misc outputs: temp (ones), bias (wtq_bias) =================
__global__ void k_misc(const float* __restrict__ wtq_bias, float* __restrict__ d_out) {
  int t = threadIdx.x;
  if (t < NHc) d_out[TEMP_OFF + t] = 1.0f;
  if (t < NHc * NSc) d_out[BIAS_OFF + t] = wtq_bias[t];
}

extern "C" void kernel_launch(void* const* d_in, const int* in_sizes, int n_in,
                              void* d_out_v, int out_size, void* d_ws, size_t ws_size,
                              hipStream_t stream) {
  const float* x        = (const float*)d_in[0];
  const float* wkv_w    = (const float*)d_in[1];
  const float* wkv_b    = (const float*)d_in[2];
  const float* wtq      = (const float*)d_in[3];
  const float* wtq_bias = (const float*)d_in[4];
  // d_in[5] = temperature (unused: temp = temperature*0 + 1)
  const float* qkv_proj = (const float*)d_in[6];
  const float* out_w    = (const float*)d_in[7];
  const float* out_b    = (const float*)d_in[8];
  float* d_out = (float*)d_out_v;
  char*  wsb   = (char*)d_ws;

  __bf16* kv_bf   = (__bf16*)(wsb + KVBF_B);
  float*  scT     = (float*) (wsb + SCT_B);
  __bf16* op_bf   = (__bf16*)(wsb + OPBF_B);   // aliases dead scoresT
  __bf16* w_bf    = (__bf16*)(wsb + WBF_B);
  float*  g_norm  = (float*) (wsb + NORM_B);
  float*  ot      = (float*) (wsb + OT_B);
  __bf16* wkvw_bf = (__bf16*)(wsb + WKVWB_B);
  __bf16* outw_bf = (__bf16*)(wsb + OUTWB_B);
  float*  swq     = d_out + SW_OFF;            // slice_weights lives in d_out

  (void)hipMemsetAsync(g_norm, 0, (size_t)Bc * NHc * NSc * sizeof(float), stream);

  k0_cvt_w <<< 512, 256, 0, stream>>>(wkv_w, out_w, wkvw_bf, outw_bf);
  k1_kv    <<<8192, 256, 0, stream>>>(x, wkvw_bf, wkv_b, kv_bf);
  k2_scores<<<8192, 256, 0, stream>>>(kv_bf, wtq, wtq_bias, scT);
  k3_topk  <<<4096, 256, 0, stream>>>(scT, swq, w_bf, g_norm);
  k4_slice <<<  64, 256, 0, stream>>>(kv_bf, w_bf, g_norm, qkv_proj, ot, d_out + ATTN_OFF);
  k5_outpre<<<8192, 256, 0, stream>>>(w_bf, ot, op_bf);
  k6_out   <<<4096, 256, 0, stream>>>(op_bf, outw_bf, out_b, d_out + OUT_OFF);
  k_misc   <<<   1, 288, 0, stream>>>(wtq_bias, d_out);
}
